// MultiHeadAttention_27797028340063
// MI455X (gfx1250) — matrix operand order repaired and checked by reading, not compiled, not run on hardware
//
#include <hip/hip_runtime.h>

// ---------------------------------------------------------------------------
// MHA forward for gfx1250 (MI455X): bf16 WMMA GEMMs + fused flash attention.
// B=8, T=1024, D=768, H=12, HD=64.
// ---------------------------------------------------------------------------

typedef __attribute__((ext_vector_type(16))) __bf16 v16bf;
typedef __attribute__((ext_vector_type(4)))  __bf16 v4bf;
typedef __attribute__((ext_vector_type(8)))  float  v8f;

#define TT   1024
#define DD   768
#define HH   12
#define HDIM 64
#define BB   8
#define MTOT (BB * TT)          // 8192
// LDS row stride in bf16 elems: 72 -> 144B rows (16B aligned for async B128
// stores; 36-bank stride is conflict-free across the 16 rows of a fragment).
#define LSTR 72

__device__ inline v8f wmma_bf16(v16bf a, v16bf b, v8f c) {
  return __builtin_amdgcn_wmma_f32_16x16x32_bf16(
      /*neg_a=*/false, a, /*neg_b=*/false, b,
      /*c_mod=*/(short)0, c, /*reuse_a=*/false, /*reuse_b=*/false);
}

union FragU { v16bf v; unsigned u[8]; };

// A fragment: 16x32 bf16, M rows striped across lanes, K split in contiguous order:
// lanes 0-15 hold K=0..15, lanes 16-31 hold K=16..31 (matches B frags).
__device__ inline v16bf load_a_frag(const __bf16* base, int stride, int row0, int k0) {
  const int lane = threadIdx.x & 31;
  const int m = lane & 15, half = lane >> 4;
  FragU f;
  const __bf16* p = base + (row0 + m) * stride + k0 + 16 * half;
#pragma unroll
  for (int j = 0; j < 8; ++j) f.u[j] = *(const unsigned*)(p + 2 * j);
  return f.v;
}

// B fragment: 32x16 bf16 with B[k][n] = S[n][k] (S row-major [n][k]):
// lanes 0-15 hold K=0..15, lanes 16-31 hold K=16..31, N = lane%16.
__device__ inline v16bf load_b_frag(const __bf16* base, int stride, int n0, int k0) {
  const int lane = threadIdx.x & 31;
  const int n = lane & 15, half = lane >> 4;
  FragU f;
  const __bf16* p = base + (n0 + n) * stride + k0 + 16 * half;
#pragma unroll
  for (int j = 0; j < 8; ++j) f.u[j] = *(const unsigned*)(p + 2 * j);
  return f.v;
}

// fp32x4 -> bf16x4 LDS store; lowers to 2x v_cvt_pk_bf16_f32 + ds_store_b64.
__device__ inline void store4bf(__bf16* dst, float4 v) {
  v4bf t;
  t[0] = (__bf16)v.x; t[1] = (__bf16)v.y;
  t[2] = (__bf16)v.z; t[3] = (__bf16)v.w;
  *(v4bf*)dst = t;
}

__device__ inline void copy16b(const __bf16* src, __bf16* dst) {
  uint4 v = *(const uint4*)src;
  ((uint2*)dst)[0] = make_uint2(v.x, v.y);
  ((uint2*)dst)[1] = make_uint2(v.z, v.w);
}

// Async global->LDS 16-byte copy (GLOBAL_LOAD_ASYNC_TO_LDS_B128, ASYNCcnt),
// falling back to a synchronous VGPR round-trip if the builtin is absent.
#if __has_builtin(__builtin_amdgcn_global_load_async_to_lds_b128)
#define HAVE_ASYNC_LDS 1
typedef __attribute__((__vector_size__(16))) int async_v4i;
typedef __attribute__((address_space(1))) async_v4i* gbl_v4i_ptr;
typedef __attribute__((address_space(3))) async_v4i* lds_v4i_ptr;
#else
#define HAVE_ASYNC_LDS 0
#endif

__device__ inline void async_copy16(const __bf16* g, __bf16* l) {
#if HAVE_ASYNC_LDS
  __builtin_amdgcn_global_load_async_to_lds_b128(
      (gbl_v4i_ptr)g, (lds_v4i_ptr)l, 0, 0);
#else
  copy16b(g, l);
#endif
}

__device__ inline void async_wait() {
#if HAVE_ASYNC_LDS
  asm volatile("s_wait_asynccnt 0" ::: "memory");
#endif
}

// ---------------------------------------------------------------------------
// Shared GEMM mainloop: C[m][n] = sum_k A[m][k] * W[n][k]   (A,W fp32 in HBM,
// converted to bf16 at LDS fill).  Block tile 128x64, 8 waves, 32x32 per wave.
// ---------------------------------------------------------------------------
__device__ inline void gemm_mainloop(const float* __restrict__ A,
                                     const float* __restrict__ W,
                                     int m0, int n0, v8f acc[2][2]) {
  __shared__ __align__(16) __bf16 As[128 * LSTR];
  __shared__ __align__(16) __bf16 Ws[64 * LSTR];
  const int tid  = threadIdx.x;
  const int wid  = tid >> 5;
  const int wm   = wid & 3;       // 4 waves along M
  const int wn   = wid >> 2;      // 2 waves along N
#pragma unroll
  for (int i = 0; i < 2; ++i)
#pragma unroll
    for (int j = 0; j < 2; ++j)
#pragma unroll
      for (int e = 0; e < 8; ++e) acc[i][j][e] = 0.0f;

  for (int kk = 0; kk < DD; kk += 64) {
    // As: 128x64 fp32 -> bf16 (2048 chunks of 4)
    for (int c = tid; c < 2048; c += 256) {
      const int row = c >> 4, seg = (c & 15) << 2;
      float4 v = *(const float4*)(A + (size_t)(m0 + row) * DD + kk + seg);
      store4bf(&As[row * LSTR + seg], v);
    }
    // Ws: 64x64 fp32 -> bf16 (1024 chunks of 4)
    for (int c = tid; c < 1024; c += 256) {
      const int row = c >> 4, seg = (c & 15) << 2;
      float4 v = *(const float4*)(W + (size_t)(n0 + row) * DD + kk + seg);
      store4bf(&Ws[row * LSTR + seg], v);
    }
    if (kk + 64 < DD) {   // global_prefetch_b8 of the next K tile
      __builtin_prefetch(A + (size_t)(m0 + (tid >> 1)) * DD + kk + 64, 0, 1);
      __builtin_prefetch(W + (size_t)(n0 + (tid & 63)) * DD + kk + 64, 0, 1);
    }
    __syncthreads();
#pragma unroll
    for (int ks = 0; ks < 2; ++ks) {
      v16bf a0 = load_a_frag(As, LSTR, wm * 32 +  0, ks * 32);
      v16bf a1 = load_a_frag(As, LSTR, wm * 32 + 16, ks * 32);
      v16bf b0 = load_b_frag(Ws, LSTR, wn * 32 +  0, ks * 32);
      v16bf b1 = load_b_frag(Ws, LSTR, wn * 32 + 16, ks * 32);
      acc[0][0] = wmma_bf16(a0, b0, acc[0][0]);
      acc[0][1] = wmma_bf16(a0, b1, acc[0][1]);
      acc[1][0] = wmma_bf16(a1, b0, acc[1][0]);
      acc[1][1] = wmma_bf16(a1, b1, acc[1][1]);
    }
    __syncthreads();
  }
}

// ---------------------------------------------------------------------------
// Kernel 1: QKV projections.  z = 0:Q  1:K  2:V(transposed per head).
// Q/K  -> bf16 [B,H,T,64];  V -> bf16 [B,H,64,T]  (so P@V B-frags are
// contiguous along K in LDS later).
// ---------------------------------------------------------------------------
__global__ __launch_bounds__(256) void qkv_kernel(
    const float* __restrict__ x,
    const float* __restrict__ Wq, const float* __restrict__ Wk,
    const float* __restrict__ Wv,
    __bf16* __restrict__ Qb, __bf16* __restrict__ Kb, __bf16* __restrict__ Vtb) {
  const int m0 = blockIdx.x * 128, n0 = blockIdx.y * 64;
  const int which = blockIdx.z;
  const float* W = (which == 0) ? Wq : (which == 1) ? Wk : Wv;
  v8f acc[2][2];
  gemm_mainloop(x, W, m0, n0, acc);

  const int tid = threadIdx.x, lane = tid & 31, wid = tid >> 5;
  const int wm = wid & 3, wn = wid >> 2;
  const int nlane = lane & 15, half = lane >> 4;
  __bf16* out = (which == 0) ? Qb : (which == 1) ? Kb : Vtb;
#pragma unroll
  for (int mt = 0; mt < 2; ++mt)
#pragma unroll
    for (int nt = 0; nt < 2; ++nt)
#pragma unroll
      for (int j = 0; j < 8; ++j) {
        const int m = m0 + wm * 32 + mt * 16 + j + 8 * half;
        const int n = n0 + wn * 32 + nt * 16 + nlane;
        const int b = m >> 10, t = m & (TT - 1);
        const int h = n >> 6,  hd = n & (HDIM - 1);
        size_t off;
        if (which == 2) off = (((size_t)(b * HH + h)) * HDIM + hd) * TT + t;
        else            off = (((size_t)(b * HH + h)) * TT + t) * HDIM + hd;
        out[off] = (__bf16)acc[mt][nt][j];
      }
}

// ---------------------------------------------------------------------------
// Kernel 2: fused causal flash attention.  One block per (q-tile of 64, b*h).
// 4 waves; wave w owns 16 query rows.  Online softmax; P re-laid-out through a
// per-wave LDS buffer (C layout -> A fragments).  ctx written fp32 [B,T,D].
// K/V/Q tiles staged with async global->LDS (ASYNCcnt) when available.
// ---------------------------------------------------------------------------
__global__ __launch_bounds__(128) void attn_kernel(
    const __bf16* __restrict__ Qb, const __bf16* __restrict__ Kb,
    const __bf16* __restrict__ Vtb, float* __restrict__ ctx) {
  __shared__ __align__(16) __bf16 Qs[64 * LSTR];
  __shared__ __align__(16) __bf16 Ks[64 * LSTR];
  __shared__ __align__(16) __bf16 Vts[64 * LSTR];
  __shared__ __align__(16) __bf16 Ps[4 * 16 * LSTR];

  const int tid = threadIdx.x, lane = tid & 31, w = tid >> 5;
  const int nlane = lane & 15, half = lane >> 4;
  const int qb = blockIdx.x;                  // q block [qb*64, qb*64+64)
  const int bh = blockIdx.y;                  // b*H + h
  const int b = bh / HH, h = bh % HH;
  const float scale = 0.125f;                 // 1/sqrt(64)
  const float LOG2E = 1.4426950408889634f;

  const size_t headQK = (size_t)bh * TT * HDIM;
  const size_t headVt = (size_t)bh * HDIM * TT;

  // stage Q tile (64 x 64 bf16) once
  {
    const __bf16* Qg = Qb + headQK + (size_t)qb * 64 * HDIM;
    for (int c = tid; c < 512; c += 128) {
      const int row = c >> 3, seg = (c & 7) * 8;
      async_copy16(Qg + row * HDIM + seg, &Qs[row * LSTR + seg]);
    }
  }

  v8f o[4];
  float mrun[8], lrun[8];
#pragma unroll
  for (int nt = 0; nt < 4; ++nt)
#pragma unroll
    for (int e = 0; e < 8; ++e) o[nt][e] = 0.0f;
#pragma unroll
  for (int j = 0; j < 8; ++j) { mrun[j] = -1e30f; lrun[j] = 0.0f; }

  const int qbase = qb * 64 + w * 16;

  for (int kb = 0; kb <= qb; ++kb) {          // causal: skip kb > qb entirely
    // stage K (64x64) and Vt (64 d x 64 kk)
    {
      const __bf16* Kg  = Kb  + headQK + (size_t)kb * 64 * HDIM;
      const __bf16* Vtg = Vtb + headVt + (size_t)kb * 64;
      for (int c = tid; c < 512; c += 128) {
        const int row = c >> 3, seg = (c & 7) * 8;
        async_copy16(Kg + row * HDIM + seg, &Ks[row * LSTR + seg]);
        async_copy16(Vtg + (size_t)row * TT + seg, &Vts[row * LSTR + seg]);
      }
    }
    async_wait();
    __syncthreads();

    // S = Q K^T  (16 q rows x 64 keys per wave)
    v8f s[4];
#pragma unroll
    for (int nt = 0; nt < 4; ++nt)
#pragma unroll
      for (int e = 0; e < 8; ++e) s[nt][e] = 0.0f;
#pragma unroll
    for (int ks = 0; ks < 2; ++ks) {
      v16bf aF = load_a_frag(Qs, LSTR, w * 16, ks * 32);
#pragma unroll
      for (int nt = 0; nt < 4; ++nt)
        s[nt] = wmma_bf16(aF, load_b_frag(Ks, LSTR, nt * 16, ks * 32), s[nt]);
    }

    // scale + causal mask + block row-max
    float mb[8];
#pragma unroll
    for (int j = 0; j < 8; ++j) mb[j] = -1e30f;
#pragma unroll
    for (int nt = 0; nt < 4; ++nt)
#pragma unroll
      for (int j = 0; j < 8; ++j) {
        float v = s[nt][j] * scale;
        const int qg = qbase + j + 8 * half;
        const int kg = kb * 64 + nt * 16 + nlane;
        if (kg > qg) v = -1e30f;
        s[nt][j] = v;
        mb[j] = fmaxf(mb[j], v);
      }
#pragma unroll
    for (int j = 0; j < 8; ++j)
      for (int xm = 1; xm < 16; xm <<= 1)
        mb[j] = fmaxf(mb[j], __shfl_xor(mb[j], xm, 32));

    // online softmax rescale
    float alpha[8], lb[8];
#pragma unroll
    for (int j = 0; j < 8; ++j) {
      const float mn = fmaxf(mrun[j], mb[j]);
      alpha[j] = exp2f((mrun[j] - mn) * LOG2E);
      mrun[j] = mn;
      lb[j] = 0.0f;
    }
#pragma unroll
    for (int nt = 0; nt < 4; ++nt)
#pragma unroll
      for (int j = 0; j < 8; ++j) {
        const float p = exp2f((s[nt][j] - mrun[j]) * LOG2E);
        s[nt][j] = p;
        lb[j] += p;
      }
#pragma unroll
    for (int j = 0; j < 8; ++j) {
      for (int xm = 1; xm < 16; xm <<= 1) lb[j] += __shfl_xor(lb[j], xm, 32);
      lrun[j] = lrun[j] * alpha[j] + lb[j];
    }
#pragma unroll
    for (int nt = 0; nt < 4; ++nt)
#pragma unroll
      for (int j = 0; j < 8; ++j) o[nt][j] *= alpha[j];

    // P (C layout, fp32) -> per-wave LDS bf16 tile -> A fragments
    __bf16* Pw = &Ps[w * 16 * LSTR];
#pragma unroll
    for (int nt = 0; nt < 4; ++nt)
#pragma unroll
      for (int j = 0; j < 8; ++j)
        Pw[(j + 8 * half) * LSTR + nt * 16 + nlane] = (__bf16)s[nt][j];
    asm volatile("s_wait_dscnt 0" ::: "memory");   // same-wave LDS RAW

    // O += P @ V   (Vt rows = d, cols = kk  ->  B[k=kk][n=d])
#pragma unroll
    for (int ks = 0; ks < 2; ++ks) {
      v16bf aF = load_a_frag(Pw, LSTR, 0, ks * 32);
#pragma unroll
      for (int nt = 0; nt < 4; ++nt)
        o[nt] = wmma_bf16(aF, load_b_frag(Vts, LSTR, nt * 16, ks * 32), o[nt]);
    }
    __syncthreads();   // protect K/Vt tiles before next fill
  }

  // normalize and write ctx fp32 [B,T,D]
#pragma unroll
  for (int nt = 0; nt < 4; ++nt)
#pragma unroll
    for (int j = 0; j < 8; ++j) {
      const float inv = 1.0f / lrun[j];
      const int t = qbase + j + 8 * half;
      const int col = h * HDIM + nt * 16 + nlane;
      ctx[((size_t)b * TT + t) * DD + col] = o[nt][j] * inv;
    }
}

// ---------------------------------------------------------------------------
// Kernel 3: output projection  out = ctx @ Wp^T + bp   (fp32 out)
// ---------------------------------------------------------------------------
__global__ __launch_bounds__(256) void proj_kernel(
    const float* __restrict__ ctx, const float* __restrict__ Wp,
    const float* __restrict__ bp, float* __restrict__ out) {
  const int m0 = blockIdx.x * 128, n0 = blockIdx.y * 64;
  v8f acc[2][2];
  gemm_mainloop(ctx, Wp, m0, n0, acc);

  const int tid = threadIdx.x, lane = tid & 31, wid = tid >> 5;
  const int wm = wid & 3, wn = wid >> 2;
  const int nlane = lane & 15, half = lane >> 4;
#pragma unroll
  for (int mt = 0; mt < 2; ++mt)
#pragma unroll
    for (int nt = 0; nt < 2; ++nt) {
      const int n = n0 + wn * 32 + nt * 16 + nlane;
      const float bias = bp[n];
#pragma unroll
      for (int j = 0; j < 8; ++j) {
        const int m = m0 + wm * 32 + mt * 16 + j + 8 * half;
        out[(size_t)m * DD + n] = acc[mt][nt][j] + bias;
      }
    }
}

// ---------------------------------------------------------------------------
extern "C" void kernel_launch(void* const* d_in, const int* in_sizes, int n_in,
                              void* d_out, int out_size, void* d_ws, size_t ws_size,
                              hipStream_t stream) {
  const float* x  = (const float*)d_in[0];
  const float* Wq = (const float*)d_in[1];
  const float* Wk = (const float*)d_in[2];
  const float* Wv = (const float*)d_in[3];
  const float* Wp = (const float*)d_in[4];
  const float* bp = (const float*)d_in[5];
  float* out = (float*)d_out;

  // workspace: Qb, Kb, Vtb (bf16, 12 MiB each) + ctx (fp32, 24 MiB)
  const size_t qkvBytes = (size_t)BB * HH * TT * HDIM * sizeof(__bf16); // 12582912
  char* wsb = (char*)d_ws;
  __bf16* Qb  = (__bf16*)(wsb);
  __bf16* Kb  = (__bf16*)(wsb + qkvBytes);
  __bf16* Vtb = (__bf16*)(wsb + 2 * qkvBytes);
  float*  ctx = (float*) (wsb + 3 * qkvBytes);

  qkv_kernel<<<dim3(MTOT / 128, DD / 64, 3), 256, 0, stream>>>(
      x, Wq, Wk, Wv, Qb, Kb, Vtb);
  attn_kernel<<<dim3(TT / 64, BB * HH), 128, 0, stream>>>(Qb, Kb, Vtb, ctx);
  proj_kernel<<<dim3(MTOT / 128, DD / 64), 256, 0, stream>>>(ctx, Wp, bp, out);
}
